// Seq2SeqWithAlignment_20358144983085
// MI455X (gfx1250) — matrix operand order repaired and checked by reading, not compiled, not run on hardware
//
#include <hip/hip_runtime.h>
#include <stdint.h>

#define ENC_MAXK 64
#define DEC_MAXT 128
#define V_SZ 32000
#define E_SZ 100
#define H_SZ 512
#define G_SZ 2048   // 4H
#define KP 128      // padded K (100 real + bias col at 100 + zeros)

#define LOG2E 1.4426950408889634f
#define LN2   0.6931471805599453f

#if __has_builtin(__builtin_amdgcn_exp2f)
#define EXP2F(x) __builtin_amdgcn_exp2f(x)
#else
#define EXP2F(x) exp2f(x)
#endif

typedef __attribute__((ext_vector_type(16))) __bf16 v16bf;
typedef __attribute__((ext_vector_type(8)))  float  v8f;

union Frag { v16bf v; uint4 q[2]; };

__device__ __forceinline__ unsigned short f2bf(float f) {
    unsigned int u = __float_as_uint(f);
    u = (u + 0x7FFFu + ((u >> 16) & 1u)) >> 16;   // RNE
    return (unsigned short)u;
}
__device__ __forceinline__ float bf2f(unsigned short u) {
    return __uint_as_float(((unsigned int)u) << 16);
}
__device__ __forceinline__ float sigf(float x){ return 1.f/(1.f+__expf(-x)); }

// ---------------------------------------------------------------- Wem -> bf16, bem folded as column 100
__global__ void k_wem_bf16(const float* __restrict__ Wem, const float* __restrict__ bem,
                           unsigned short* __restrict__ WembB) {
    int v = blockIdx.x*blockDim.x + threadIdx.x;
    if (v >= V_SZ) return;
    const float* src = Wem + (size_t)v*E_SZ;
    unsigned short* dst = WembB + (size_t)v*KP;
    #pragma unroll 4
    for (int e=0;e<E_SZ;e++) dst[e] = f2bf(src[e]);
    dst[E_SZ] = f2bf(bem[v]);
    for (int e=E_SZ+1;e<KP;e++) dst[e]=0;
}

// ---------------------------------------------------------------- pre[t][g] = Wih@emb[token] + bih + bhh
__global__ void k_pregates(const int* __restrict__ x, const int* __restrict__ y,
                           const float* __restrict__ emb,
                           const float* __restrict__ eWih, const float* __restrict__ eBih, const float* __restrict__ eBhh,
                           const float* __restrict__ dWih, const float* __restrict__ dBih, const float* __restrict__ dBhh,
                           float* __restrict__ preE, float* __restrict__ preD) {
    __shared__ float xe[E_SZ];
    int b = blockIdx.x;
    bool enc = b < ENC_MAXK;
    int t = enc ? b : b - ENC_MAXK;
    int tok = enc ? x[t] : y[t];
    const float* Wih = enc ? eWih : dWih;
    const float* bih = enc ? eBih : dBih;
    const float* bhh = enc ? eBhh : dBhh;
    float* out = (enc ? preE : preD) + (size_t)t*G_SZ;
    for (int e=threadIdx.x; e<E_SZ; e+=blockDim.x) xe[e] = emb[(size_t)tok*E_SZ + e];
    __syncthreads();
    for (int g=threadIdx.x; g<G_SZ; g+=blockDim.x) {
        const float* wr = Wih + (size_t)g*E_SZ;
        float acc = bih[g] + bhh[g];
        #pragma unroll 4
        for (int e=0;e<E_SZ;e++) acc = fmaf(wr[e], xe[e], acc);
        out[g] = acc;
    }
}

// ---------------------------------------------------------------- sequential LSTMs (block0=enc, block1=dec)
__global__ void __launch_bounds__(1024)
k_lstm(const float* __restrict__ preE, const float* __restrict__ preD,
       const float* __restrict__ eWhh, const float* __restrict__ dWhh,
       float* __restrict__ Eenc, float* __restrict__ Dst) {
    __shared__ float h[H_SZ];
    __shared__ float gates[G_SZ];
    bool enc = (blockIdx.x == 0);
    int T = enc ? ENC_MAXK : DEC_MAXT;
    const float* pre = enc ? preE : preD;
    const float* Whh = enc ? eWhh : dWhh;
    float* out = enc ? Eenc : Dst;
    int tid = threadIdx.x;
    float c = 0.f;
    for (int j=tid;j<H_SZ;j+=1024) h[j]=0.f;
    __syncthreads();
    for (int t=0;t<T;t++) {
        const float* pt = pre + (size_t)t*G_SZ;
        #pragma unroll
        for (int gi=0; gi<2; gi++) {
            int g = tid + gi*1024;
            const float* wr = Whh + (size_t)g*H_SZ;
            float acc = pt[g];
            #pragma unroll 8
            for (int j=0;j<H_SZ;j++) acc = fmaf(wr[j], h[j], acc);
            gates[g]=acc;
        }
        __syncthreads();
        if (tid < H_SZ) {
            float ig = sigf(gates[tid]);
            float fg = sigf(gates[H_SZ+tid]);
            float gg = tanhf(gates[2*H_SZ+tid]);
            float og = sigf(gates[3*H_SZ+tid]);
            c = fg*c + ig*gg;
            float hh = og*tanhf(c);
            h[tid]=hh;
            out[(size_t)t*H_SZ + tid]=hh;
        }
        __syncthreads();
    }
}

// ---------------------------------------------------------------- EWE = Eenc@We_E^T, DWD = D@We_D^T
__global__ void k_proj(const float* __restrict__ Eenc, const float* __restrict__ Dst,
                       const float* __restrict__ Weh,
                       float* __restrict__ EWE, float* __restrict__ DWD) {
    int b = blockIdx.x, e = threadIdx.x;
    if (e >= E_SZ) return;
    if (b < ENC_MAXK) {
        const float* hr = Eenc + (size_t)b*H_SZ;
        const float* wr = Weh + (size_t)e*(2*H_SZ);
        float acc=0.f;
        #pragma unroll 8
        for(int j=0;j<H_SZ;j++) acc=fmaf(hr[j],wr[j],acc);
        EWE[(size_t)b*KP+e]=acc;
    } else {
        int t = b-ENC_MAXK;
        const float* hr = Dst + (size_t)t*H_SZ;
        const float* wr = Weh + (size_t)e*(2*H_SZ) + H_SZ;
        float acc=0.f;
        #pragma unroll 8
        for(int j=0;j<H_SZ;j++) acc=fmaf(hr[j],wr[j],acc);
        DWD[(size_t)t*KP+e]=acc;
    }
}

// ---------------------------------------------------------------- hpre (bf16, padded, PRE-SCALED by log2(e))
// Scaling the whole A matrix by log2(e) puts the GEMM output in the log2
// domain: logsumexp pass needs only a bare v_exp_f32 (exp2) per element.
__global__ void k_hpre(const float* __restrict__ EWE, const float* __restrict__ DWD,
                       const float* __restrict__ beh, unsigned short* __restrict__ hpreB) {
    int t = blockIdx.x;
    for (int idx=threadIdx.x; idx<ENC_MAXK*KP; idx+=blockDim.x) {
        int k = idx>>7, e = idx&(KP-1);
        float v;
        if (e < E_SZ) {
            v = EWE[(size_t)k*KP+e] + DWD[(size_t)t*KP+e] + beh[e];
            v = (v > 0.f ? v : 0.f) * LOG2E;
        } else v = (e==E_SZ) ? LOG2E : 0.f;
        hpreB[((size_t)t*ENC_MAXK + k)*KP + e] = f2bf(v);
    }
}

// ---------------------------------------------------------------- WMMA GEMM + two-pass logsumexp + gather
// block = (t, mtile): 16 rows of hpre vs all 32000 vocab rows; 8 waves split N.
// Pass 1: row max only (v_max). Pass 2: recompute WMMA, sum exp2(acc - MX).
__global__ void __launch_bounds__(256)
k_emission(const unsigned short* __restrict__ hpreB, const unsigned short* __restrict__ WembB,
           const int* __restrict__ y, float* __restrict__ emission) {
    __shared__ unsigned short Alds[16*KP];     // 4 KB A-tile (bf16, log2-scaled)
    __shared__ float sred[8][16];              // per-wave partial max / sum
    __shared__ float rowmax[16];               // final row max (log2 domain)
    int blk = blockIdx.x;
    int t = blk >> 2, mt = blk & 3;
    int m0 = t*ENC_MAXK + mt*16;
    int tid = threadIdx.x;

    // stage A tile (16 x 128 bf16 = 256 uint4) into LDS
    reinterpret_cast<uint4*>(Alds)[tid] =
        reinterpret_cast<const uint4*>(hpreB + (size_t)m0*KP)[tid];
    __syncthreads();

    int w = tid >> 5, lane = tid & 31;
    int Mrow = lane & 15, khalf = lane >> 4;

    // A fragments per ISA layout: halves 0..7 = K base..base+7, 8..15 = K 16+base.., base = 8*khalf
    Frag af[4];
    #pragma unroll
    for (int kb=0; kb<4; kb++) {
        const unsigned short* ap = Alds + Mrow*KP + kb*32 + khalf*8;
        af[kb].q[0] = *reinterpret_cast<const uint4*>(ap);
        af[kb].q[1] = *reinterpret_cast<const uint4*>(ap + 16);
    }

    const int NT = V_SZ/16;                    // 2000 N-tiles

    // ---------------- pass 1: row max (branch-free, WMMA-bound) ----------------
    float mx[8];
    #pragma unroll
    for (int r=0;r<8;r++) mx[r] = -3.4e38f;

    for (int nt = w; nt < NT; nt += 8) {
        int vbase = nt*16;
        // unconditional prefetch with clamped address: no exec-mask branch
        int pv = (nt + 8 < NT) ? vbase + 128 : vbase;
        __builtin_prefetch(WembB + (size_t)(pv + Mrow)*KP, 0, 3);
        v8f acc = {};
        #pragma unroll
        for (int kb=0; kb<4; kb++) {
            Frag bf;
            const unsigned short* bp = WembB + (size_t)(vbase + Mrow)*KP + kb*32 + khalf*8;
            bf.q[0] = *reinterpret_cast<const uint4*>(bp);
            bf.q[1] = *reinterpret_cast<const uint4*>(bp + 16);
            acc = __builtin_amdgcn_wmma_f32_16x16x32_bf16(
                      false, af[kb].v, false, bf.v, (short)0, acc, false, false);
        }
        #pragma unroll
        for (int r=0;r<8;r++) mx[r] = fmaxf(mx[r], acc[r]);
    }
    #pragma unroll
    for (int d=1; d<16; d<<=1)
        #pragma unroll
        for (int r=0;r<8;r++) mx[r] = fmaxf(mx[r], __shfl_xor(mx[r], d, 32));
    if ((lane & 15) == 0) {
        #pragma unroll
        for (int r=0;r<8;r++) sred[w][khalf*8 + r] = mx[r];
    }
    __syncthreads();
    if (tid < 16) {
        float m = sred[0][tid];
        #pragma unroll
        for (int ww=1; ww<8; ww++) m = fmaxf(m, sred[ww][tid]);
        rowmax[tid] = m;
    }
    __syncthreads();
    float mxr[8];
    #pragma unroll
    for (int r=0;r<8;r++) mxr[r] = rowmax[khalf*8 + r];
    __syncthreads();   // sred reused below

    // ---------------- pass 2: sum of exp2 (one v_exp per logit) ----------------
    float sm[8];
    #pragma unroll
    for (int r=0;r<8;r++) sm[r] = 0.f;

    for (int nt = w; nt < NT; nt += 8) {
        int vbase = nt*16;
        int pv = (nt + 8 < NT) ? vbase + 128 : vbase;
        __builtin_prefetch(WembB + (size_t)(pv + Mrow)*KP, 0, 3);
        v8f acc = {};
        #pragma unroll
        for (int kb=0; kb<4; kb++) {
            Frag bf;
            const unsigned short* bp = WembB + (size_t)(vbase + Mrow)*KP + kb*32 + khalf*8;
            bf.q[0] = *reinterpret_cast<const uint4*>(bp);
            bf.q[1] = *reinterpret_cast<const uint4*>(bp + 16);
            acc = __builtin_amdgcn_wmma_f32_16x16x32_bf16(
                      false, af[kb].v, false, bf.v, (short)0, acc, false, false);
        }
        #pragma unroll
        for (int r=0;r<8;r++) sm[r] += EXP2F(acc[r] - mxr[r]);
    }
    #pragma unroll
    for (int d=1; d<16; d<<=1)
        #pragma unroll
        for (int r=0;r<8;r++) sm[r] += __shfl_xor(sm[r], d, 32);
    if ((lane & 15) == 0) {
        #pragma unroll
        for (int r=0;r<8;r++) sred[w][khalf*8 + r] = sm[r];
    }
    __syncthreads();

    if (tid < 16) {
        int row = tid;
        float s = sred[0][row];
        #pragma unroll
        for (int ww=1; ww<8; ww++) s += sred[ww][row];
        float lse2 = rowmax[row] + log2f(s);            // log2-domain logsumexp
        int word = y[t+1];
        const unsigned short* wr = WembB + (size_t)word*KP;   // includes bem at col 100
        const unsigned short* ar = Alds + row*KP;             // log2-scaled hpre row
        float dot = 0.f;
        #pragma unroll 8
        for (int e=0;e<KP;e++) dot = fmaf(bf2f(ar[e]), bf2f(wr[e]), dot);
        emission[m0 + row] = (dot - lse2) * LN2;              // back to natural log
    }
}

// ---------------------------------------------------------------- Viterbi DP + backtrace
__global__ void k_viterbi(const float* __restrict__ emission, int* __restrict__ inds,
                          int* __restrict__ outAlign, float* __restrict__ outVal) {
    __shared__ float alpha[2][ENC_MAXK];
    const float fs = -0.40546510810816444f;  // log(128/192)
    const float fe = -1.0986122886681098f;   // log(64/192)
    int j = threadIdx.x;
    alpha[0][j] = 0.f;
    __syncthreads();
    int cur = 0;
    for (int t=0; t<DEC_MAXT-1; t++) {
        int c = ENC_MAXK-1 - j;               // flipped column
        float best = -3.4e38f; int bi = 0;
        #pragma unroll 8
        for (int i=0;i<ENC_MAXK;i++) {
            float tr = (i + c < ENC_MAXK) ? (float)(ENC_MAXK-1-(i+c))*fs + fe : -3.4e38f;
            float s = alpha[cur][i] + tr;
            if (s > best) { best = s; bi = i; }
        }
        alpha[1-cur][j] = best + emission[t*ENC_MAXK + j];
        inds[t*ENC_MAXK + j] = bi;
        __syncthreads();
        cur = 1-cur;
    }
    if (j == 0) {
        *outVal = alpha[cur][ENC_MAXK-1];
        int ind = ENC_MAXK-1;
        for (int tt=DEC_MAXT-1; tt>=1; tt--) {
            ind = inds[(tt-1)*ENC_MAXK + ind];
            outAlign[tt] = ind;
        }
        outAlign[0] = 0;                       // indices[0] is the zeros row
    }
}

// ---------------------------------------------------------------- host launcher
extern "C" void kernel_launch(void* const* d_in, const int* in_sizes, int n_in,
                              void* d_out, int out_size, void* d_ws, size_t ws_size,
                              hipStream_t stream) {
    const int*   x       = (const int*)  d_in[0];
    const int*   y       = (const int*)  d_in[1];
    const float* emb     = (const float*)d_in[2];
    const float* enc_Wih = (const float*)d_in[3];
    const float* enc_Whh = (const float*)d_in[4];
    const float* enc_bih = (const float*)d_in[5];
    const float* enc_bhh = (const float*)d_in[6];
    const float* dec_Wih = (const float*)d_in[7];
    const float* dec_Whh = (const float*)d_in[8];
    const float* dec_bih = (const float*)d_in[9];
    const float* dec_bhh = (const float*)d_in[10];
    const float* Weh     = (const float*)d_in[11];
    const float* beh     = (const float*)d_in[12];
    const float* Wem     = (const float*)d_in[13];
    const float* bem     = (const float*)d_in[14];

    char* p = (char*)d_ws;
    auto alloc = [&](size_t bytes)->void* {
        void* r = (void*)p;
        p += (bytes + 255) & ~(size_t)255;
        return r;
    };
    unsigned short* WembB = (unsigned short*)alloc((size_t)V_SZ*KP*2);           // 8.2 MB
    float* preE  = (float*)alloc((size_t)ENC_MAXK*G_SZ*4);
    float* preD  = (float*)alloc((size_t)DEC_MAXT*G_SZ*4);
    float* Eenc  = (float*)alloc((size_t)ENC_MAXK*H_SZ*4);
    float* Dst   = (float*)alloc((size_t)DEC_MAXT*H_SZ*4);
    float* EWE   = (float*)alloc((size_t)ENC_MAXK*KP*4);
    float* DWD   = (float*)alloc((size_t)(DEC_MAXT-1)*KP*4);
    unsigned short* hpreB = (unsigned short*)alloc((size_t)(DEC_MAXT-1)*ENC_MAXK*KP*2); // 2.1 MB
    float* emis  = (float*)alloc((size_t)(DEC_MAXT-1)*ENC_MAXK*4);
    int*   inds  = (int*)  alloc((size_t)(DEC_MAXT-1)*ENC_MAXK*4);

    int* outAlign = (int*)d_out;
    float* outVal = (float*)d_out + DEC_MAXT;

    k_wem_bf16 <<<V_SZ/256, 256, 0, stream>>>(Wem, bem, WembB);
    k_pregates <<<ENC_MAXK + DEC_MAXT, 256, 0, stream>>>(x, y, emb,
                 enc_Wih, enc_bih, enc_bhh, dec_Wih, dec_bih, dec_bhh, preE, preD);
    k_lstm     <<<2, 1024, 0, stream>>>(preE, preD, enc_Whh, dec_Whh, Eenc, Dst);
    k_proj     <<<ENC_MAXK + (DEC_MAXT-1), 128, 0, stream>>>(Eenc, Dst, Weh, EWE, DWD);
    k_hpre     <<<DEC_MAXT-1, 256, 0, stream>>>(EWE, DWD, beh, hpreB);
    k_emission <<<(DEC_MAXT-1)*4, 256, 0, stream>>>(hpreB, WembB, y, emis);
    k_viterbi  <<<1, ENC_MAXK, 0, stream>>>(emis, inds, outAlign, outVal);
}
